// ImprovedGeometricTrunk_9818295239420
// MI455X (gfx1250) — matrix operand order repaired and checked by reading, not compiled, not run on hardware
//
#include <hip/hip_runtime.h>
#include <math.h>

typedef __attribute__((ext_vector_type(16))) _Float16 v16h;
typedef __attribute__((ext_vector_type(8)))  _Float16 v8h;
typedef __attribute__((ext_vector_type(8)))  float    v8f;

#define NREFS 10
#define H 128
#define P 64
#define TILE_M 16
#define WAVES_PER_BLOCK 4
#define BLOCK_THREADS (WAVES_PER_BLOCK * 32)
#define M_TOTAL (32 * 16384)
#define N_TILES (M_TOTAL / TILE_M)
#define GRID_BLOCKS 2048

#define W1F_HALVES (8 * 512)        // 8 col-tiles x (32 lanes x 16 halves), K padded 13->32
#define W2F_HALVES (8 * 4 * 512)    // 8 col-tiles x 4 K-chunks
#define W3F_HALVES (4 * 4 * 512)    // 4 col-tiles x 4 K-chunks
#define REF_OFF_HALVES (W1F_HALVES + W2F_HALVES + W3F_HALVES)

#define PI_F 3.14159265358979323846f
#define EPS_F 1e-5f
#define SH_STRIDE 144               // 128 + 16 halves: 288B rows (16B aligned, bank-shifted)

// sched_group_barrier masks: 0x8 = MFMA/WMMA, 0x100 = DS read
#if defined(__has_builtin)
#if __has_builtin(__builtin_amdgcn_sched_group_barrier)
#define SGB(mask, n) __builtin_amdgcn_sched_group_barrier((mask), (n), 0)
#endif
#endif
#ifndef SGB
#define SGB(mask, n)
#endif

// ---------------------------------------------------------------------------
// Prep: build f16 WMMA B-fragments for W1 (K-padded to 32), W2, W3 in the
// workspace, plus cos/sin/phi of the reference points.
// B (KxN, f16, 16x16x32): lanes 0-15 hold col n = lane, K = h<8 ? h : h+8;
// lanes 16-31 hold col n = lane-16, K = 8 + (h<8 ? h : h+8).
// ---------------------------------------------------------------------------
__global__ void geo_prep_kernel(const float* __restrict__ W1,
                                const float* __restrict__ W2,
                                const float* __restrict__ W3,
                                const float* __restrict__ ref_theta,
                                const float* __restrict__ ref_phi,
                                _Float16* __restrict__ wsf,
                                float* __restrict__ wsref) {
  const int tid = blockIdx.x * blockDim.x + threadIdx.x;
  const int stride = gridDim.x * blockDim.x;

  for (int i = tid; i < W1F_HALVES; i += stride) {
    int j = i >> 9, r = i & 511, lane = r >> 4, h = r & 15;
    int k = ((lane < 16) ? 0 : 8) + ((h < 8) ? h : h + 8);
    int c = j * 16 + (lane & 15);
    wsf[i] = (k < NREFS + 3) ? (_Float16)W1[k * H + c] : (_Float16)0.0f;
  }
  for (int i = tid; i < W2F_HALVES; i += stride) {
    int t = i >> 9, r = i & 511, lane = r >> 4, h = r & 15;
    int j = t >> 2, kk = t & 3;
    int k = kk * 32 + ((lane < 16) ? 0 : 8) + ((h < 8) ? h : h + 8);
    int c = j * 16 + (lane & 15);
    wsf[W1F_HALVES + i] = (_Float16)W2[k * H + c];
  }
  for (int i = tid; i < W3F_HALVES; i += stride) {
    int t = i >> 9, r = i & 511, lane = r >> 4, h = r & 15;
    int j = t >> 2, kk = t & 3;
    int k = kk * 32 + ((lane < 16) ? 0 : 8) + ((h < 8) ? h : h + 8);
    int c = j * 16 + (lane & 15);
    wsf[W1F_HALVES + W2F_HALVES + i] = (_Float16)W3[k * P + c];
  }
  if (tid < NREFS) {
    float t = ref_theta[tid];
    wsref[tid]             = __cosf(t);
    wsref[NREFS + tid]     = __sinf(t);
    wsref[2 * NREFS + tid] = ref_phi[tid];
  }
}

// ---------------------------------------------------------------------------
// Main fused kernel (persistent blocks): weight fragments live in LDS; each
// wave grid-strides over 16-point tiles. B-fragment ds_loads run 2 fragments
// ahead of their WMMA, pinned with sched_group_barrier.
// ---------------------------------------------------------------------------
__global__ __launch_bounds__(BLOCK_THREADS)
void geo_trunk_kernel(const float* __restrict__ coords,
                      const _Float16* __restrict__ wsf,
                      const float* __restrict__ wsref,
                      const float* __restrict__ b1, const float* __restrict__ g1,
                      const float* __restrict__ bt1,
                      const float* __restrict__ b2, const float* __restrict__ g2,
                      const float* __restrict__ bt2,
                      const float* __restrict__ b3,
                      float* __restrict__ out) {
  __shared__ __align__(32) _Float16 sW[REF_OFF_HALVES];
  __shared__ __align__(32) _Float16 sH[WAVES_PER_BLOCK][TILE_M][SH_STRIDE];
  __shared__ float sBias[6 * H + P + 3 * NREFS];

  // ---- one-time cooperative LDS fill (fragments are pre-swizzled: memcpy)
  {
    const v8h* src = (const v8h*)wsf;
    v8h* dst = (v8h*)sW;
    for (int i = threadIdx.x; i < REF_OFF_HALVES / 8; i += BLOCK_THREADS)
      dst[i] = src[i];
    for (int i = threadIdx.x; i < H; i += BLOCK_THREADS) {
      sBias[0 * H + i] = b1[i];  sBias[1 * H + i] = g1[i];  sBias[2 * H + i] = bt1[i];
      sBias[3 * H + i] = b2[i];  sBias[4 * H + i] = g2[i];  sBias[5 * H + i] = bt2[i];
    }
    if (threadIdx.x < P) sBias[6 * H + threadIdx.x] = b3[threadIdx.x];
    if (threadIdx.x < 3 * NREFS) sBias[6 * H + P + threadIdx.x] = wsref[threadIdx.x];
  }
  __syncthreads();

  const float* sB1  = &sBias[0 * H];
  const float* sG1  = &sBias[1 * H];
  const float* sBt1 = &sBias[2 * H];
  const float* sB2  = &sBias[3 * H];
  const float* sG2  = &sBias[4 * H];
  const float* sBt2 = &sBias[5 * H];
  const float* sB3  = &sBias[6 * H];
  const float* sRef = &sBias[6 * H + P];

  const int lane  = threadIdx.x & 31;
  const int wave  = threadIdx.x >> 5;
  const int m     = lane & 15;            // A-row / B-col / D-col slot
  const int abase = (lane < 16) ? 0 : 8;  // A K-interleave base; also D row base
  const int wstep = GRID_BLOCKS * WAVES_PER_BLOCK;

  // B-fragment loader (32 contiguous bytes per lane -> 2x ds_load_b128)
  auto ldW = [&](int halfoff) -> v16h {
    return *(const v16h*)(sW + halfoff + lane * 16);
  };

  for (int tile = blockIdx.x * WAVES_PER_BLOCK + wave; tile < N_TILES; tile += wstep) {
    const int row0 = tile * TILE_M;

    { // prefetch the next tile's coords while we compute this one
      int nrow = (tile + wstep) * TILE_M;
      if (nrow < M_TOTAL) __builtin_prefetch(&coords[(size_t)nrow * 2], 0, 1);
    }

    // ---- features: A fragment (16x32 f16). Lane m holds row m; halves
    // 0..7 -> K = abase+0..7, halves 8..15 -> K = 16+abase+0..7.
    float theta = coords[(row0 + m) * 2 + 0];
    float phi   = coords[(row0 + m) * 2 + 1];
    float ct = __cosf(theta), st = __sinf(theta);

    v16h a1;
    #pragma unroll
    for (int h = 0; h < 16; ++h) {
      int k = abase + ((h < 8) ? h : h + 8);
      float f = 0.0f;
      if (k < NREFS) {
        float cd = ct * sRef[k] + st * sRef[NREFS + k] * __cosf(phi - sRef[2 * NREFS + k]);
        cd = fminf(1.0f, fmaxf(-1.0f, cd));
        f = acosf(cd) * (1.0f / PI_F);
      } else if (k == NREFS)     f = theta * (1.0f / PI_F);
      else if (k == NREFS + 1)   f = phi * (1.0f / (2.0f * PI_F));
      else if (k == NREFS + 2)   f = 1.0f;
      a1[h] = (_Float16)f;
    }

    // ---- GEMM1: [16x32] x [32x128] -> 8 D tiles; depth-2 pipelined B loads
    v8f c1[8];
    {
      v16h bb0 = ldW(0);
      v16h bb1 = ldW(512);
      #pragma unroll
      for (int j = 0; j < 8; ++j) {
        v16h bn = (j + 2 < 8) ? ldW((j + 2) * 512) : bb0;
        v8f z = {};
        c1[j] = __builtin_amdgcn_wmma_f32_16x16x32_f16(false, a1, false, bb0,
                                                       (short)0, z, false, false);
        bb0 = bb1; bb1 = bn;
      }
      // schedule: 2 frags (4 ds reads) ahead, then (wmma, 2 ds reads) x6, wmma x2
      SGB(0x100, 4);
      #pragma unroll
      for (int t = 0; t < 6; ++t) { SGB(0x8, 1); SGB(0x100, 2); }
      SGB(0x8, 2);
    }

    // ---- LayerNorm1 + ReLU -> sH.  D layout: reg r, lanes 0-15: row r,
    // col = lane; lanes 16-31: row r+8, col = lane-16.
    {
      float ps[8], pq[8];
      #pragma unroll
      for (int r = 0; r < 8; ++r) { ps[r] = 0.0f; pq[r] = 0.0f; }
      #pragma unroll
      for (int j = 0; j < 8; ++j) {
        float bb = sB1[j * 16 + m];
        #pragma unroll
        for (int r = 0; r < 8; ++r) {
          float v = c1[j][r] + bb;
          c1[j][r] = v;
          ps[r] += v; pq[r] += v * v;
        }
      }
      #pragma unroll
      for (int d = 1; d < 16; d <<= 1) {
        #pragma unroll
        for (int r = 0; r < 8; ++r) {
          ps[r] += __shfl_xor(ps[r], d, 32);
          pq[r] += __shfl_xor(pq[r], d, 32);
        }
      }
      float mu[8], is[8];
      #pragma unroll
      for (int r = 0; r < 8; ++r) {
        mu[r] = ps[r] * (1.0f / H);
        float var = pq[r] * (1.0f / H) - mu[r] * mu[r];
        is[r] = rsqrtf(var + EPS_F);
      }
      #pragma unroll
      for (int j = 0; j < 8; ++j) {
        float gg = sG1[j * 16 + m], bb = sBt1[j * 16 + m];
        #pragma unroll
        for (int r = 0; r < 8; ++r) {
          float v = (c1[j][r] - mu[r]) * is[r] * gg + bb;
          v = fmaxf(v, 0.0f);
          sH[wave][abase + r][j * 16 + m] = (_Float16)v;
        }
      }
    }

    // ---- A fragments for GEMM2 (two contiguous v8h runs per 32-K chunk)
    v16h a2[4];
    #pragma unroll
    for (int kk = 0; kk < 4; ++kk) {
      v8h lo = *(const v8h*)&sH[wave][m][kk * 32 + abase];
      v8h hi = *(const v8h*)&sH[wave][m][kk * 32 + abase + 16];
      #pragma unroll
      for (int h = 0; h < 8; ++h) { a2[kk][h] = lo[h]; a2[kk][h + 8] = hi[h]; }
    }

    // ---- GEMM2: [16x128] x [128x128]; flattened (j,kk), depth-2 pipeline
    v8f c2[8];
    {
      v16h bb0 = ldW(W1F_HALVES + 0);
      v16h bb1 = ldW(W1F_HALVES + 512);
      #pragma unroll
      for (int t = 0; t < 32; ++t) {
        v16h bn = (t + 2 < 32) ? ldW(W1F_HALVES + (t + 2) * 512) : bb0;
        int j = t >> 2, kk = t & 3;
        if (kk == 0) {
          v8f z = {};
          c2[j] = __builtin_amdgcn_wmma_f32_16x16x32_f16(false, a2[kk], false, bb0,
                                                         (short)0, z, false, false);
        } else {
          c2[j] = __builtin_amdgcn_wmma_f32_16x16x32_f16(false, a2[kk], false, bb0,
                                                         (short)0, c2[j], false, false);
        }
        bb0 = bb1; bb1 = bn;
      }
      // schedule: A frags (8 reads) + 2 B frags (4 reads) ahead,
      // then (wmma, 2 ds reads) x30, wmma x2
      SGB(0x100, 8);
      SGB(0x100, 4);
      #pragma unroll
      for (int t = 0; t < 30; ++t) { SGB(0x8, 1); SGB(0x100, 2); }
      SGB(0x8, 2);
    }

    // ---- LayerNorm2 + ReLU -> sH
    {
      float ps[8], pq[8];
      #pragma unroll
      for (int r = 0; r < 8; ++r) { ps[r] = 0.0f; pq[r] = 0.0f; }
      #pragma unroll
      for (int j = 0; j < 8; ++j) {
        float bb = sB2[j * 16 + m];
        #pragma unroll
        for (int r = 0; r < 8; ++r) {
          float v = c2[j][r] + bb;
          c2[j][r] = v;
          ps[r] += v; pq[r] += v * v;
        }
      }
      #pragma unroll
      for (int d = 1; d < 16; d <<= 1) {
        #pragma unroll
        for (int r = 0; r < 8; ++r) {
          ps[r] += __shfl_xor(ps[r], d, 32);
          pq[r] += __shfl_xor(pq[r], d, 32);
        }
      }
      float mu[8], is[8];
      #pragma unroll
      for (int r = 0; r < 8; ++r) {
        mu[r] = ps[r] * (1.0f / H);
        float var = pq[r] * (1.0f / H) - mu[r] * mu[r];
        is[r] = rsqrtf(var + EPS_F);
      }
      #pragma unroll
      for (int j = 0; j < 8; ++j) {
        float gg = sG2[j * 16 + m], bb = sBt2[j * 16 + m];
        #pragma unroll
        for (int r = 0; r < 8; ++r) {
          float v = (c2[j][r] - mu[r]) * is[r] * gg + bb;
          v = fmaxf(v, 0.0f);
          sH[wave][abase + r][j * 16 + m] = (_Float16)v;
        }
      }
    }

    // ---- A fragments for GEMM3
    v16h a3[4];
    #pragma unroll
    for (int kk = 0; kk < 4; ++kk) {
      v8h lo = *(const v8h*)&sH[wave][m][kk * 32 + abase];
      v8h hi = *(const v8h*)&sH[wave][m][kk * 32 + abase + 16];
      #pragma unroll
      for (int h = 0; h < 8; ++h) { a3[kk][h] = lo[h]; a3[kk][h + 8] = hi[h]; }
    }

    // ---- GEMM3: [16x128] x [128x64] + b3 -> out; depth-2 pipeline
    {
      v8f c3[4];
      v16h bb0 = ldW(W1F_HALVES + W2F_HALVES + 0);
      v16h bb1 = ldW(W1F_HALVES + W2F_HALVES + 512);
      #pragma unroll
      for (int t = 0; t < 16; ++t) {
        v16h bn = (t + 2 < 16) ? ldW(W1F_HALVES + W2F_HALVES + (t + 2) * 512) : bb0;
        int j = t >> 2, kk = t & 3;
        if (kk == 0) {
          v8f z = {};
          c3[j] = __builtin_amdgcn_wmma_f32_16x16x32_f16(false, a3[kk], false, bb0,
                                                         (short)0, z, false, false);
        } else {
          c3[j] = __builtin_amdgcn_wmma_f32_16x16x32_f16(false, a3[kk], false, bb0,
                                                         (short)0, c3[j], false, false);
        }
        bb0 = bb1; bb1 = bn;
      }
      // schedule: A frags (8 reads) + 2 B frags (4 reads), (wmma, 2 reads) x14, wmma x2
      SGB(0x100, 8);
      SGB(0x100, 4);
      #pragma unroll
      for (int t = 0; t < 14; ++t) { SGB(0x8, 1); SGB(0x100, 2); }
      SGB(0x8, 2);

      #pragma unroll
      for (int j = 0; j < 4; ++j) {
        float bb = sB3[j * 16 + m];
        #pragma unroll
        for (int r = 0; r < 8; ++r) {
          out[(size_t)(row0 + abase + r) * P + j * 16 + m] = c3[j][r] + bb;
        }
      }
    }
  }
}

extern "C" void kernel_launch(void* const* d_in, const int* in_sizes, int n_in,
                              void* d_out, int out_size, void* d_ws, size_t ws_size,
                              hipStream_t stream) {
  const float* coords    = (const float*)d_in[0];
  const float* ref_theta = (const float*)d_in[1];
  const float* ref_phi   = (const float*)d_in[2];
  const float* W1  = (const float*)d_in[3];
  const float* b1  = (const float*)d_in[4];
  const float* g1  = (const float*)d_in[5];
  const float* bt1 = (const float*)d_in[6];
  const float* W2  = (const float*)d_in[7];
  const float* b2  = (const float*)d_in[8];
  const float* g2  = (const float*)d_in[9];
  const float* bt2 = (const float*)d_in[10];
  const float* W3  = (const float*)d_in[11];
  const float* b3  = (const float*)d_in[12];

  _Float16* wsf = (_Float16*)d_ws;
  float* wsref  = (float*)((char*)d_ws + (size_t)REF_OFF_HALVES * sizeof(_Float16));

  geo_prep_kernel<<<64, 128, 0, stream>>>(W1, W2, W3, ref_theta, ref_phi, wsf, wsref);

  geo_trunk_kernel<<<GRID_BLOCKS, BLOCK_THREADS, 0, stream>>>(
      coords, wsf, wsref, b1, g1, bt1, b2, g2, bt2, b3, (float*)d_out);
}
